// METG_8409545965693
// MI455X (gfx1250) — matrix operand order repaired and checked by reading, not compiled, not run on hardware
//
#include <hip/hip_runtime.h>
#include <math.h>

// ---------------------------------------------------------------------------
// CDNA5 (gfx1250) types
// ---------------------------------------------------------------------------
typedef __attribute__((ext_vector_type(16))) __bf16 v16bf;
typedef __attribute__((ext_vector_type(8)))  __bf16 v8bf;
typedef __attribute__((ext_vector_type(8)))  float  v8f;

__device__ __forceinline__ __bf16 f2bf(float x) { return (__bf16)x; }

#define EPI_RELU 1
#define EPI_RES  2
#define EPI_PE   4

#define BM 64
#define BN 128
#define BK 32
#define PAD 8      // 8 bf16 = 16B row padding: keeps 16B-aligned vector LDS stores

#define LN10000 9.210340371976184f

// ---------------------------------------------------------------------------
// Generic bf16-WMMA GEMM:  C[M,N] = epi( A[M,K] @ W^T + bias )
//   A addressing:  addr = (r >> aShift)*aBS + (r & ((1<<aShift)-1))*aRS + k*aCS
//   W addressing:  addr = n*wRS + k*wCS
// 8 waves per block, 64x128 macro tile, each wave 32x32 (2x2 WMMA tiles).
// Fast vectorized staging (b128 loads + b128 LDS stores) when K-stride == 1.
// ---------------------------------------------------------------------------
__global__ __launch_bounds__(256)
void gemm_bf16_wmma(const float* __restrict__ A, const float* __restrict__ W,
                    const float* __restrict__ bias, const float* __restrict__ res,
                    float* __restrict__ C,
                    int M, int N, int K,
                    long aBS, int aShift, int aRS, int aCS,
                    int wRS, int wCS,
                    int cLD, int resLD,
                    int flags, int peS, int peD)
{
    __shared__ __bf16 As[BM][BK + PAD];
    __shared__ __bf16 Ws[BN][BK + PAD];

    const int tid  = threadIdx.x;
    const int lane = tid & 31;
    const int wave = tid >> 5;
    const int mr   = lane & 15;
    const int g    = lane >> 4;

    const int bm = blockIdx.y * BM;
    const int bn = blockIdx.x * BN;
    const int wm = (wave >> 2) * 32;   // 0 / 32
    const int wn = (wave & 3) * 32;    // 0..96

    const bool fastA = (aCS == 1);
    const bool fastW = (wCS == 1);
    const int  aMask = (1 << aShift) - 1;

    v8f acc[2][2];
    #pragma unroll
    for (int a0 = 0; a0 < 2; a0++)
        #pragma unroll
        for (int b0 = 0; b0 < 2; b0++)
            #pragma unroll
            for (int i = 0; i < 8; i++) acc[a0][b0][i] = 0.0f;

    for (int k0 = 0; k0 < K; k0 += BK) {
        // ---- stage A tile (fp32 -> bf16) ----
        if (fastA) {
            int row = tid >> 2;               // 0..63
            int cg  = (tid & 3) * 8;          // 0,8,16,24
            int gr  = bm + row;
            v8bf pv;
            if (gr < M) {
                int rb = gr >> aShift;
                int rr = gr & aMask;
                const float* ap = A + (long)rb * aBS + (long)rr * aRS + (k0 + cg);
                float4 a0 = *(const float4*)ap;
                float4 a1 = *(const float4*)(ap + 4);
                pv[0] = f2bf(a0.x); pv[1] = f2bf(a0.y); pv[2] = f2bf(a0.z); pv[3] = f2bf(a0.w);
                pv[4] = f2bf(a1.x); pv[5] = f2bf(a1.y); pv[6] = f2bf(a1.z); pv[7] = f2bf(a1.w);
            } else {
                #pragma unroll
                for (int e = 0; e < 8; e++) pv[e] = f2bf(0.0f);
            }
            *(v8bf*)&As[row][cg] = pv;
        } else {
            for (int e = tid; e < BM * BK; e += 256) {
                int r = e >> 5, c = e & 31;
                int gr = bm + r, gk = k0 + c;
                float v = 0.0f;
                if (gr < M && gk < K) {
                    int rb = gr >> aShift;
                    int rr = gr & aMask;
                    v = A[(long)rb * aBS + (long)rr * aRS + (long)gk * aCS];
                }
                As[r][c] = f2bf(v);
            }
        }
        // ---- stage W tile (fp32 -> bf16) ----
        if (fastW) {
            int row = tid >> 1;               // 0..127
            int cg  = (tid & 1) * 16;         // 0,16
            int gn  = bn + row;
            v8bf p0, p1;
            if (gn < N) {
                const float* wp = W + (long)gn * wRS + (k0 + cg);
                float4 w0 = *(const float4*)wp;
                float4 w1 = *(const float4*)(wp + 4);
                float4 w2 = *(const float4*)(wp + 8);
                float4 w3 = *(const float4*)(wp + 12);
                p0[0] = f2bf(w0.x); p0[1] = f2bf(w0.y); p0[2] = f2bf(w0.z); p0[3] = f2bf(w0.w);
                p0[4] = f2bf(w1.x); p0[5] = f2bf(w1.y); p0[6] = f2bf(w1.z); p0[7] = f2bf(w1.w);
                p1[0] = f2bf(w2.x); p1[1] = f2bf(w2.y); p1[2] = f2bf(w2.z); p1[3] = f2bf(w2.w);
                p1[4] = f2bf(w3.x); p1[5] = f2bf(w3.y); p1[6] = f2bf(w3.z); p1[7] = f2bf(w3.w);
            } else {
                #pragma unroll
                for (int e = 0; e < 8; e++) { p0[e] = f2bf(0.0f); p1[e] = f2bf(0.0f); }
            }
            *(v8bf*)&Ws[row][cg]     = p0;
            *(v8bf*)&Ws[row][cg + 8] = p1;
        } else {
            for (int e = tid; e < BN * BK; e += 256) {
                int r = e >> 5, c = e & 31;
                int gn = bn + r, gk = k0 + c;
                float v = 0.0f;
                if (gn < N && gk < K) v = W[(long)gn * wRS + (long)gk * wCS];
                Ws[r][c] = f2bf(v);
            }
        }
        __syncthreads();

        if (k0 + BK < K)  // global_prefetch_b8 hint for next weight tile
            __builtin_prefetch(W + (long)bn * wRS + (long)(k0 + BK) * wCS, 0, 1);

        // A fragment: lane = row (mr), elems j / j+8  <->  K = g*8+j / g*8+16+j
        v16bf af[2];
        #pragma unroll
        for (int s = 0; s < 2; s++) {
            int ml = wm + s * 16 + mr;
            #pragma unroll
            for (int j = 0; j < 8; j++) {
                af[s][j]     = As[ml][g * 8 + j];
                af[s][j + 8] = As[ml][g * 8 + 16 + j];
            }
        }
        // B fragment: lane = col (mr), elem e <-> K = g*8+e (e<8) / g*8+8+e (e>=8)
        v16bf bfr[2];
        #pragma unroll
        for (int s = 0; s < 2; s++) {
            int nl = wn + s * 16 + mr;
            #pragma unroll
            for (int e = 0; e < 16; e++) bfr[s][e] = Ws[nl][(e < 8) ? g * 8 + e : g * 8 + 8 + e];
        }

        #pragma unroll
        for (int sm = 0; sm < 2; sm++)
            #pragma unroll
            for (int sn = 0; sn < 2; sn++)
                acc[sm][sn] = __builtin_amdgcn_wmma_f32_16x16x32_bf16(
                    false, af[sm], false, bfr[sn], (short)0, acc[sm][sn], false, false);
        __syncthreads();
    }

    // epilogue (C layout: lane = col, VGPR i -> row i + 8*g)
    #pragma unroll
    for (int sm = 0; sm < 2; sm++)
        #pragma unroll
        for (int sn = 0; sn < 2; sn++)
            #pragma unroll
            for (int i = 0; i < 8; i++) {
                int row = bm + wm + sm * 16 + i + 8 * g;
                int col = bn + wn + sn * 16 + mr;
                if (row < M && col < N) {
                    float v = acc[sm][sn][i];
                    if (bias) v += bias[col];
                    if (flags & EPI_RES) v += res[(long)row * resLD + col];
                    if (flags & EPI_PE) {
                        float ang = (float)(row % peS) *
                                    __expf((float)(col & ~1) * (-LN10000 / (float)peD));
                        v += (col & 1) ? __cosf(ang) : __sinf(ang);
                    }
                    if (flags & EPI_RELU) v = fmaxf(v, 0.0f);
                    C[(long)row * cLD + col] = v;
                }
            }
}

// ---------------------------------------------------------------------------
// Flash attention: one wave per 16-row q-tile, online softmax over S keys.
// qkv rows = [q | k | v] (1536 floats), head dim 64, scale 1/8 folded into Q.
// ---------------------------------------------------------------------------
__global__ __launch_bounds__(128)
void attn_flash(const float* __restrict__ qkv, float* __restrict__ o,
                int Bb, int S, int Hh, int D3, int Dm)
{
    __shared__ __bf16 Ps[4][16][34];
    const int lane = threadIdx.x & 31;
    const int wv   = threadIdx.x >> 5;
    const int mr   = lane & 15;
    const int g    = lane >> 4;
    const int dh   = 64;

    int tile = blockIdx.x * 4 + wv;
    int ntq  = S >> 4;
    int qt   = tile % ntq;
    int bh   = tile / ntq;
    int h    = bh % Hh;
    int b    = bh / Hh;

    // Q as A-fragment, pre-scaled by rsqrt(dh)=0.125
    v16bf qf[2];
    {
        long qrow = (long)(b * S + qt * 16 + mr) * D3 + h * dh;
        #pragma unroll
        for (int p = 0; p < 2; p++)
            #pragma unroll
            for (int j = 0; j < 8; j++) {
                qf[p][j]     = f2bf(qkv[qrow + p * 32 + g * 16 + j] * 0.125f);
                qf[p][j + 8] = f2bf(qkv[qrow + p * 32 + g * 16 + 8 + j] * 0.125f);
            }
    }

    v8f ov[4];
    #pragma unroll
    for (int t = 0; t < 4; t++)
        #pragma unroll
        for (int i = 0; i < 8; i++) ov[t][i] = 0.0f;
    float rm[8], rl[8];
    #pragma unroll
    for (int i = 0; i < 8; i++) { rm[i] = -1e30f; rl[i] = 0.0f; }

    for (int kt = 0; kt < (S >> 5); kt++) {
        // scores for 32 keys (two 16x16 tiles), K = dh = 64 -> 2 WMMAs each
        v8f sc[2];
        #pragma unroll
        for (int sub = 0; sub < 2; sub++) {
            const float* kp = qkv + (long)(b * S + kt * 32 + sub * 16 + mr) * D3 + Dm + h * dh + g * 16;
            v16bf kf0, kf1;
            #pragma unroll
            for (int e = 0; e < 16; e++) {      // B layout: elem e <-> K = g*16+e
                kf0[e] = f2bf(kp[e]);
                kf1[e] = f2bf(kp[32 + e]);
            }
            v8f z;
            #pragma unroll
            for (int i = 0; i < 8; i++) z[i] = 0.0f;
            z = __builtin_amdgcn_wmma_f32_16x16x32_bf16(false, qf[0], false, kf0, (short)0, z, false, false);
            sc[sub] = __builtin_amdgcn_wmma_f32_16x16x32_bf16(false, qf[1], false, kf1, (short)0, z, false, false);
        }

        // online softmax (row stats per C-layout row i+8g; reduce across 16 lanes)
        #pragma unroll
        for (int i = 0; i < 8; i++) {
            float m0 = fmaxf(sc[0][i], sc[1][i]);
            #pragma unroll
            for (int off = 8; off >= 1; off >>= 1) m0 = fmaxf(m0, __shfl_xor(m0, off, 32));
            float mn = fmaxf(rm[i], m0);
            float fr = __expf(rm[i] - mn);
            float p0 = __expf(sc[0][i] - mn);
            float p1 = __expf(sc[1][i] - mn);
            sc[0][i] = p0; sc[1][i] = p1;
            float s = p0 + p1;
            #pragma unroll
            for (int off = 8; off >= 1; off >>= 1) s += __shfl_xor(s, off, 32);
            rl[i] = rl[i] * fr + s;
            rm[i] = mn;
            #pragma unroll
            for (int t = 0; t < 4; t++) ov[t][i] *= fr;
        }

        // transpose P (C layout) -> A layout through LDS
        __syncthreads();
        #pragma unroll
        for (int sub = 0; sub < 2; sub++)
            #pragma unroll
            for (int i = 0; i < 8; i++)
                Ps[wv][i + 8 * g][sub * 16 + mr] = f2bf(sc[sub][i]);
        __syncthreads();

        v16bf pf;
        #pragma unroll
        for (int j = 0; j < 8; j++) {
            pf[j]     = Ps[wv][mr][g * 16 + j];
            pf[j + 8] = Ps[wv][mr][g * 16 + 8 + j];
        }

        // P(16x32) @ V(32x64) -> 4 output tiles
        #pragma unroll
        for (int t = 0; t < 4; t++) {
            v16bf vf;
            int n = t * 16 + mr;
            #pragma unroll
            for (int e = 0; e < 16; e++)
                vf[e] = f2bf(qkv[(long)(b * S + kt * 32 + g * 16 + e) * D3 + 2 * Dm + h * dh + n]);
            ov[t] = __builtin_amdgcn_wmma_f32_16x16x32_bf16(false, pf, false, vf, (short)0, ov[t], false, false);
        }
    }

    #pragma unroll
    for (int t = 0; t < 4; t++)
        #pragma unroll
        for (int i = 0; i < 8; i++) {
            int row = b * S + qt * 16 + i + 8 * g;
            int col = h * dh + t * 16 + mr;
            o[(long)row * Dm + col] = ov[t][i] / rl[i];
        }
}

// ---------------------------------------------------------------------------
// Row-wise LayerNorm (one wave per row)
// ---------------------------------------------------------------------------
__global__ __launch_bounds__(256)
void layernorm_rows(const float* __restrict__ x, const float* __restrict__ gam,
                    const float* __restrict__ bet, float* __restrict__ y,
                    int M, int N)
{
    int row  = blockIdx.x * 8 + (threadIdx.x >> 5);
    int lane = threadIdx.x & 31;
    if (row >= M) return;
    const float* xr = x + (long)row * N;
    float s = 0.0f;
    for (int c = lane; c < N; c += 32) s += xr[c];
    #pragma unroll
    for (int o = 16; o >= 1; o >>= 1) s += __shfl_xor(s, o, 32);
    float mean = s / (float)N;
    float v = 0.0f;
    for (int c = lane; c < N; c += 32) { float d = xr[c] - mean; v += d * d; }
    #pragma unroll
    for (int o = 16; o >= 1; o >>= 1) v += __shfl_xor(v, o, 32);
    float r = rsqrtf(v / (float)N + 1e-5f);
    float* yr = y + (long)row * N;
    for (int c = lane; c < N; c += 32) yr[c] = (xr[c] - mean) * r * gam[c] + bet[c];
}

// Row-wise softmax, in place (one wave per row)
__global__ __launch_bounds__(256)
void softmax_rows(float* __restrict__ x, int M, int N)
{
    int row  = blockIdx.x * 8 + (threadIdx.x >> 5);
    int lane = threadIdx.x & 31;
    if (row >= M) return;
    float* xr = x + (long)row * N;
    float m = -1e30f;
    for (int c = lane; c < N; c += 32) m = fmaxf(m, xr[c]);
    #pragma unroll
    for (int o = 16; o >= 1; o >>= 1) m = fmaxf(m, __shfl_xor(m, o, 32));
    float s = 0.0f;
    for (int c = lane; c < N; c += 32) { float e = __expf(xr[c] - m); xr[c] = e; s += e; }
    #pragma unroll
    for (int o = 16; o >= 1; o >>= 1) s += __shfl_xor(s, o, 32);
    float inv = 1.0f / s;
    for (int c = lane; c < N; c += 32) xr[c] *= inv;
}

// ---------------------------------------------------------------------------
// Graph branch helpers (tiny compute; plain fp32)
// ---------------------------------------------------------------------------
__global__ __launch_bounds__(256)
void zero_f32(float* p, int n) { int i = blockIdx.x * 256 + threadIdx.x; if (i < n) p[i] = 0.0f; }

__global__ __launch_bounds__(256)
void feat_mean(const float* __restrict__ x, float* __restrict__ f, int Bb, int S, int F)
{
    int i = blockIdx.x * 256 + threadIdx.x;
    if (i >= Bb * F) return;
    int b = i / F, c = i % F;
    float s = 0.0f;
    for (int t = 0; t < S; t++) s += x[(long)b * S * F + (long)t * F + c];
    f[i] = s / (float)S;
}

__global__ __launch_bounds__(32)
void feat_norm(const float* __restrict__ f, float* __restrict__ fn, int Bb, int F)
{
    int b = threadIdx.x;
    if (b >= Bb) return;
    float s = 0.0f;
    for (int c = 0; c < F; c++) { float v = f[b * F + c]; s += v * v; }
    float inv = 1.0f / fmaxf(sqrtf(s), 1e-12f);
    for (int c = 0; c < F; c++) fn[b * F + c] = f[b * F + c] * inv;
}

// top-4 of sim[b,i,:]=fn[b,i]*fn[b,:], drop argmax, mark 3 neighbors
__global__ __launch_bounds__(256)
void topk_adj(const float* __restrict__ fn, float* __restrict__ A, int Bb, int F)
{
    int i = blockIdx.x * 256 + threadIdx.x;
    if (i >= Bb * F) return;
    int b = i / F, r = i % F;
    const float* v = fn + b * F;
    float fi = v[r];
    float bv[4] = {-1e30f, -1e30f, -1e30f, -1e30f};
    int   bi[4] = {-1, -1, -1, -1};
    for (int j = 0; j < F; j++) {
        float s = fi * v[j];
        #pragma unroll
        for (int t = 0; t < 4; t++) {
            if (s > bv[t]) {          // strict > keeps lowest index on ties
                for (int u = 3; u > t; u--) { bv[u] = bv[u - 1]; bi[u] = bi[u - 1]; }
                bv[t] = s; bi[t] = j; break;
            }
        }
    }
    for (int t = 1; t < 4; t++)
        if (bi[t] >= 0) A[(long)b * F * F + (long)r * F + bi[t]] = 1.0f;
}

// symmetrize + self-loop + degree-normalize
__global__ __launch_bounds__(256)
void adj_norm(const float* __restrict__ A, float* __restrict__ An, int Bb, int F)
{
    int i = blockIdx.x * 256 + threadIdx.x;
    if (i >= Bb * F) return;
    int b = i / F, r = i % F;
    const float* Ab = A + (long)b * F * F;
    float deg = 0.0f;
    for (int j = 0; j < F; j++)
        deg += (j == r) ? 1.0f : fmaxf(Ab[r * F + j], Ab[j * F + r]);
    float inv = 1.0f / deg;
    float* Ao = An + (long)b * F * F + (long)r * F;
    for (int j = 0; j < F; j++)
        Ao[j] = ((j == r) ? 1.0f : fmaxf(Ab[r * F + j], Ab[j * F + r])) * inv;
}

// out[b,i,c] = sum_j An[b,i,j] * T[b*F+j, c]  (optional relu / transposed store)
__global__ __launch_bounds__(256)
void graph_apply(const float* __restrict__ An, const float* __restrict__ T,
                 float* __restrict__ out, int Bb, int F, int N,
                 int do_relu, int transpose_store)
{
    long idx = (long)blockIdx.x * 256 + threadIdx.x;
    if (idx >= (long)Bb * F * N) return;
    int c = (int)(idx % N);
    long r = idx / N;
    int i = (int)(r % F);
    int b = (int)(r / F);
    const float* a  = An + (long)b * F * F + (long)i * F;
    const float* tb = T + (long)b * F * N;
    float acc = 0.0f;
    for (int j = 0; j < F; j++) acc += a[j] * tb[(long)j * N + c];
    if (do_relu) acc = fmaxf(acc, 0.0f);
    if (transpose_store) out[(long)b * N * F + (long)c * F + i] = acc;
    else                 out[r * N + c] = acc;
}

// ---------------------------------------------------------------------------
// Host orchestration
// ---------------------------------------------------------------------------
static inline void launch_gemm(hipStream_t stream,
    const float* A, const float* W, const float* bias, const float* res, float* C,
    int M, int N, int K, long aBS, int aShift, int aRS, int aCS,
    int wRS, int wCS, int cLD, int resLD, int flags, int peS, int peD)
{
    dim3 grid((N + BN - 1) / BN, (M + BM - 1) / BM);
    gemm_bf16_wmma<<<grid, dim3(256), 0, stream>>>(A, W, bias, res, C, M, N, K,
        aBS, aShift, aRS, aCS, wRS, wCS, cLD, resLD, flags, peS, peD);
}

extern "C" void kernel_launch(void* const* d_in, const int* in_sizes, int n_in,
                              void* d_out, int out_size, void* d_ws, size_t ws_size,
                              hipStream_t stream)
{
    (void)in_sizes; (void)n_in; (void)out_size; (void)ws_size;
    enum { Bz = 32, Sz = 512, Fz = 64, Dz = 512, Fff = 2048, Mz = 1024, Lz = 3, Hz = 8 };
    const int Mtok = Bz * Sz;                       // 16384 tokens
    const int NOBATCH = 30;                         // aShift sentinel: r>>30 == 0

    const float* x     = (const float*)d_in[0];
    const float* W_in  = (const float*)d_in[1];
    const float* b_in  = (const float*)d_in[2];
    const float* qkv_w = (const float*)d_in[3];
    const float* qkv_b = (const float*)d_in[4];
    const float* out_w = (const float*)d_in[5];
    const float* out_b = (const float*)d_in[6];
    const float* ln1_g = (const float*)d_in[7];
    const float* ln1_b = (const float*)d_in[8];
    const float* ln2_g = (const float*)d_in[9];
    const float* ln2_b = (const float*)d_in[10];
    const float* ff1_w = (const float*)d_in[11];
    const float* ff1_b = (const float*)d_in[12];
    const float* ff2_w = (const float*)d_in[13];
    const float* ff2_b = (const float*)d_in[14];
    const float* memw  = (const float*)d_in[15];
    const float* fc1_w = (const float*)d_in[16];
    const float* fc1_b = (const float*)d_in[17];
    const float* fc2_w = (const float*)d_in[18];
    const float* fc2_b = (const float*)d_in[19];
    const float* gc1_w = (const float*)d_in[20];
    const float* gc1_b = (const float*)d_in[21];
    const float* gc2_w = (const float*)d_in[22];
    const float* gc2_b = (const float*)d_in[23];

    float* out   = (float*)d_out;
    float* t_out = out;                                   // 32*512*64
    float* g_out = out + (size_t)Bz * Sz * Fz;            // 32*512*64
    float* attnb = g_out + (size_t)Bz * Sz * Fz;          // 32*512*1024

    // workspace layout (~192 MiB)
    float* h   = (float*)d_ws;                            // 16384*512
    float* o   = h   + (size_t)Mtok * Dz;                 // 16384*512
    float* tmp = o   + (size_t)Mtok * Dz;                 // 16384*512
    float* big = tmp + (size_t)Mtok * Dz;                 // 16384*1536 (qkv / ff / scratch)

    // 1) input projection + sinusoidal positional encoding
    launch_gemm(stream, x, W_in, b_in, nullptr, h, Mtok, Dz, Fz,
                0L, NOBATCH, Fz, 1, Fz, 1, Dz, 0, EPI_PE, Sz, Dz);

    // 2) transformer layers
    for (int l = 0; l < Lz; l++) {
        const float* qw  = qkv_w + (size_t)l * 3 * Dz * Dz;
        const float* qb  = qkv_b + (size_t)l * 3 * Dz;
        const float* ow  = out_w + (size_t)l * Dz * Dz;
        const float* ob  = out_b + (size_t)l * Dz;
        const float* f1w = ff1_w + (size_t)l * Fff * Dz;
        const float* f1b = ff1_b + (size_t)l * Fff;
        const float* f2w = ff2_w + (size_t)l * Dz * Fff;
        const float* f2b = ff2_b + (size_t)l * Dz;

        launch_gemm(stream, h, qw, qb, nullptr, big, Mtok, 3 * Dz, Dz,
                    0L, NOBATCH, Dz, 1, Dz, 1, 3 * Dz, 0, 0, 0, 1);
        attn_flash<<<dim3(Bz * Hz * (Sz / 16) / 4), dim3(128), 0, stream>>>(
            big, o, Bz, Sz, Hz, 3 * Dz, Dz);
        launch_gemm(stream, o, ow, ob, h, tmp, Mtok, Dz, Dz,
                    0L, NOBATCH, Dz, 1, Dz, 1, Dz, Dz, EPI_RES, 0, 1);
        layernorm_rows<<<dim3(Mtok / 8), dim3(256), 0, stream>>>(
            tmp, ln1_g + (size_t)l * Dz, ln1_b + (size_t)l * Dz, h, Mtok, Dz);

        const int CH = 4096;                  // chunk FF to bound scratch
        for (int cs = 0; cs < Mtok; cs += CH) {
            launch_gemm(stream, h + (size_t)cs * Dz, f1w, f1b, nullptr, big,
                        CH, Fff, Dz, 0L, NOBATCH, Dz, 1, Dz, 1, Fff, 0, EPI_RELU, 0, 1);
            launch_gemm(stream, big, f2w, f2b, h + (size_t)cs * Dz, tmp + (size_t)cs * Dz,
                        CH, Dz, Fff, 0L, NOBATCH, Fff, 1, Fff, 1, Dz, Dz, EPI_RES, 0, 1);
        }
        layernorm_rows<<<dim3(Mtok / 8), dim3(256), 0, stream>>>(
            tmp, ln2_g + (size_t)l * Dz, ln2_b + (size_t)l * Dz, h, Mtok, Dz);
    }

    // 3) memory attention — sims straight into the attn output slot
    launch_gemm(stream, h, memw, nullptr, nullptr, attnb, Mtok, Mz, Dz,
                0L, NOBATCH, Dz, 1, Dz, 1, Mz, 0, 0, 0, 1);
    softmax_rows<<<dim3(Mtok / 8), dim3(256), 0, stream>>>(attnb, Mtok, Mz);
    // upd = attn @ memory  (W accessed transposed: wRS=1, wCS=Dz)
    launch_gemm(stream, attnb, memw, nullptr, nullptr, o, Mtok, Dz, Mz,
                0L, NOBATCH, Mz, 1, 1, Dz, Dz, 0, 0, 0, 1);

    // 4) head: relu([h, upd] @ fc1^T + b) @ fc2^T + b  (split-K concat)
    launch_gemm(stream, h, fc1_w, nullptr, nullptr, tmp, Mtok, Dz, Dz,
                0L, NOBATCH, Dz, 1, 2 * Dz, 1, Dz, 0, 0, 0, 1);
    launch_gemm(stream, o, fc1_w + Dz, fc1_b, tmp, big, Mtok, Dz, Dz,
                0L, NOBATCH, Dz, 1, 2 * Dz, 1, Dz, Dz, EPI_RES | EPI_RELU, 0, 1);
    launch_gemm(stream, big, fc2_w, fc2_b, nullptr, t_out, Mtok, Fz, Dz,
                0L, NOBATCH, Dz, 1, Dz, 1, Fz, 0, 0, 0, 1);

    // 5) graph branch (reuses `big` scratch after fc2 consumed it)
    float* gf  = big;
    float* gfn = big + Bz * Fz;
    float* gA  = big + 2 * Bz * Fz;
    float* gAn = gA + (size_t)Bz * Fz * Fz;
    float* gt  = gAn + (size_t)Bz * Fz * Fz;
    float* gh  = gt + (size_t)Bz * Fz * Dz;
    float* gt2 = gh + (size_t)Bz * Fz * Dz;

    feat_mean<<<dim3((Bz * Fz + 255) / 256), dim3(256), 0, stream>>>(x, gf, Bz, Sz, Fz);
    feat_norm<<<dim3(1), dim3(32), 0, stream>>>(gf, gfn, Bz, Fz);
    zero_f32<<<dim3((Bz * Fz * Fz + 255) / 256), dim3(256), 0, stream>>>(gA, Bz * Fz * Fz);
    topk_adj<<<dim3((Bz * Fz + 255) / 256), dim3(256), 0, stream>>>(gfn, gA, Bz, Fz);
    adj_norm<<<dim3((Bz * Fz + 255) / 256), dim3(256), 0, stream>>>(gA, gAn, Bz, Fz);

    // xt @ gc1_w^T + b1  (A = x read transposed: 64 rows per batch -> shift 6)
    launch_gemm(stream, x, gc1_w, gc1_b, nullptr, gt, Bz * Fz, Dz, Sz,
                (long)Sz * Fz, 6, 1, Fz, Sz, 1, Dz, 0, 0, 0, 1);
    graph_apply<<<dim3((Bz * Fz * Dz + 255) / 256), dim3(256), 0, stream>>>(
        gAn, gt, gh, Bz, Fz, Dz, 1, 0);
    launch_gemm(stream, gh, gc2_w, gc2_b, nullptr, gt2, Bz * Fz, Sz, Dz,
                0L, NOBATCH, Dz, 1, Dz, 1, Sz, 0, 0, 0, 1);
    graph_apply<<<dim3((Bz * Fz * Sz + 255) / 256), dim3(256), 0, stream>>>(
        gAn, gt2, g_out, Bz, Fz, Sz, 0, 1);
}
